// StudentModel_45938970198313
// MI455X (gfx1250) — compile-verified, matrix operand order checked by reading, not run on hardware
//
#include <hip/hip_runtime.h>
#include <hip/hip_fp16.h>

// ---------------------------------------------------------------------------
// CDNA5 (gfx1250) pose-net. Heavy op = seg 3x3 conv over bilinear-upsampled
// feats (~0.97 TFLOP): implicit-GEMM with v_wmma_f32_16x16x32_f16, on-the-fly
// bilinear interpolation from a 258KB LDS activation block, and double-buffered
// async global->LDS weight staging (ASYNCcnt) when the builtin is available.
// ---------------------------------------------------------------------------

typedef __attribute__((ext_vector_type(16))) _Float16 v16h;
typedef __attribute__((ext_vector_type(8)))  _Float16 v8h;
typedef __attribute__((ext_vector_type(8)))  float    v8f;

#if __has_builtin(__builtin_amdgcn_global_load_async_to_lds_b128) && \
    __has_builtin(__builtin_amdgcn_s_wait_asynccnt)
#define SEG_ASYNC_LDS 1
// builtin expects pointers to 16B int vectors in AS1 (global) / AS3 (LDS)
typedef int v4i_vs __attribute__((__vector_size__(16)));
typedef __attribute__((address_space(1))) v4i_vs* as1_b128;
typedef __attribute__((address_space(3))) v4i_vs* as3_b128;
#else
#define SEG_ASYNC_LDS 0
#endif

// ------------------------------- backbone ----------------------------------

__global__ void stem_kernel(const float* __restrict__ x, const float* __restrict__ w,
                            const float* __restrict__ sc, const float* __restrict__ sh,
                            float* __restrict__ out) {
  int idx = blockIdx.x * blockDim.x + threadIdx.x;          // 32*128*128
  if (idx >= 32 * 128 * 128) return;
  int ox = idx & 127, oy = (idx >> 7) & 127, co = idx >> 14;
  float acc = 0.f;
  for (int ci = 0; ci < 3; ++ci)
    for (int ky = 0; ky < 3; ++ky) {
      int iy = oy * 2 + ky - 1;
      if ((unsigned)iy >= 256u) continue;
      for (int kx = 0; kx < 3; ++kx) {
        int ix = ox * 2 + kx - 1;
        if ((unsigned)ix >= 256u) continue;
        acc += x[(ci * 256 + iy) * 256 + ix] * w[((co * 3 + ci) * 3 + ky) * 3 + kx];
      }
    }
  float v = acc * sc[co] + sh[co];
  out[idx] = fminf(fmaxf(v, 0.f), 6.f);
}

__global__ void conv1x1_kernel(const float* __restrict__ in, const float* __restrict__ w,
                               const float* __restrict__ sc, const float* __restrict__ sh,
                               const float* __restrict__ residual, float* __restrict__ out,
                               int Cin, int Cout, int HW, int relu6) {
  int idx = blockIdx.x * blockDim.x + threadIdx.x;
  if (idx >= Cout * HW) return;
  int hw = idx % HW, co = idx / HW;
  const float* wr = w + (size_t)co * Cin;
  float acc = 0.f;
  for (int ci = 0; ci < Cin; ++ci) acc += wr[ci] * in[(size_t)ci * HW + hw];
  acc = acc * sc[co] + sh[co];
  if (relu6) acc = fminf(fmaxf(acc, 0.f), 6.f);
  if (residual) acc += residual[idx];
  out[idx] = acc;
}

__global__ void dwconv_kernel(const float* __restrict__ in, const float* __restrict__ w,
                              const float* __restrict__ sc, const float* __restrict__ sh,
                              float* __restrict__ out, int C, int Hin, int Hout, int stride) {
  int idx = blockIdx.x * blockDim.x + threadIdx.x;
  if (idx >= C * Hout * Hout) return;
  int ox = idx % Hout, oy = (idx / Hout) % Hout, c = idx / (Hout * Hout);
  float acc = 0.f;
  for (int ky = 0; ky < 3; ++ky) {
    int iy = oy * stride + ky - 1;
    if ((unsigned)iy >= (unsigned)Hin) continue;
    for (int kx = 0; kx < 3; ++kx) {
      int ix = ox * stride + kx - 1;
      if ((unsigned)ix >= (unsigned)Hin) continue;
      acc += in[((size_t)c * Hin + iy) * Hin + ix] * w[c * 9 + ky * 3 + kx];
    }
  }
  float v = acc * sc[c] + sh[c];
  out[idx] = fminf(fmaxf(v, 0.f), 6.f);
}

// feats fp32 CHW [1280][64]  ->  fp16 NHWC [64][1280]
__global__ void feats_half_kernel(const float* __restrict__ f, _Float16* __restrict__ fh) {
  int idx = blockIdx.x * blockDim.x + threadIdx.x;          // [hw][c]
  if (idx >= 64 * 1280) return;
  int c = idx % 1280, hw = idx / 1280;
  fh[idx] = (_Float16)f[c * 64 + hw];
}

// hm / paf: 1x1 conv + bias (tiny)
__global__ void proj_kernel(const float* __restrict__ f, const float* __restrict__ w,
                            const float* __restrict__ b, float* __restrict__ out, int K) {
  int idx = blockIdx.x * blockDim.x + threadIdx.x;
  if (idx >= K * 64) return;
  int hw = idx & 63, k = idx >> 6;
  const float* wr = w + (size_t)k * 1280;
  float acc = b[k];
  for (int c = 0; c < 1280; ++c) acc += wr[c] * f[c * 64 + hw];
  out[idx] = acc;
}

__global__ void peaks_kernel(const float* __restrict__ hm, float* __restrict__ out) {
  int idx = blockIdx.x * blockDim.x + threadIdx.x;          // 21*64
  if (idx >= 21 * 64) return;
  int x = idx & 7, y = (idx >> 3) & 7;
  float r = 0.f;
  if (x >= 1 && x <= 6 && y >= 1 && y <= 6) {
    float c = hm[idx];
    if (c > 0.1f && c >= hm[idx - 8] && c >= hm[idx + 8] &&
        c >= hm[idx - 1] && c >= hm[idx + 1]) r = 1.f;
  }
  out[idx] = r;
}

// seg_w1 fp32 OIHW [640][1280][3][3] -> fp16 chunked GEMM-B layout:
//   wt[(kk9*40 + cc)][n(640)][k(32)]   with k = channel-within-chunk
// so per-chunk staging into LDS is a pure contiguous copy (async-LDS friendly).
__global__ void prep_wt_kernel(const float* __restrict__ w1, _Float16* __restrict__ wt) {
  size_t idx = (size_t)blockIdx.x * blockDim.x + threadIdx.x;
  if (idx >= (size_t)9 * 40 * 640 * 32) return;
  int k   = (int)(idx & 31);
  int n   = (int)((idx >> 5) % 640);
  int blk = (int)(idx / (640 * 32));
  int cc  = blk % 40, kk9 = blk / 40;
  int c   = cc * 32 + k;
  int ky  = kk9 / 3, kx = kk9 % 3;
  wt[idx] = (_Float16)w1[(((size_t)n * 1280 + c) * 3 + ky) * 3 + kx];
}

// ------------------------- seg WMMA implicit GEMM --------------------------
// Tile: 8x8 output pixels per workgroup (grid 32x32). 256 threads = 8 wave32:
// wave = (msub 0..3) x (npair 0..1): 16 px x 64 ch -> 4 accumulators, 4 WMMA
// per K-chunk. N loop: 5 tiles of 128 channels. K = 9*1280 in K=32 chunks.
// B tiles double-buffered in LDS with a 16B-chunk rotate swizzle (rot=(n>>2)&3)
// so every fragment read is a conflict-free ds_load_b128 per 16-lane phase.

#define USTRIDE 1288   // 1280 + pad -> conflict-free A-fragment b128 reads
#define SEG_SMEM_BYTES (100 * USTRIDE * 2 + 2 * 128 * 32 * 2 + 64 * 4)

__launch_bounds__(256)
__global__ void seg_wmma_kernel(const _Float16* __restrict__ featsH,   // [64][1280]
                                const _Float16* __restrict__ wt,       // [360][640][32]
                                const float* __restrict__ bn_scale,
                                const float* __restrict__ bn_shift,
                                const float* __restrict__ w2,
                                const float* __restrict__ b2,
                                float* __restrict__ seg_out) {
  extern __shared__ char smem_raw[];
  _Float16* U   = (_Float16*)smem_raw;                 // [100][USTRIDE]
  _Float16* Bs0 = U + 100 * USTRIDE;                   // [128][32]
  _Float16* Bs1 = Bs0 + 128 * 32;                      // [128][32]
  float* seg_acc = (float*)(Bs1 + 128 * 32);           // [64]

  const int tid = threadIdx.x;
  const int Y0 = (blockIdx.x >> 5) * 8;
  const int X0 = (blockIdx.x & 31) * 8;

  if (tid < 64) seg_acc[tid] = 0.f;

  // ---- Build bilinear-interpolated activation block: 10x10 positions x 1280ch
  for (int pos = 0; pos < 100; ++pos) {
    int uy = Y0 - 1 + pos / 10;
    int ux = X0 - 1 + pos % 10;
    _Float16* Urow = U + pos * USTRIDE;
    if ((unsigned)uy >= 256u || (unsigned)ux >= 256u) {   // conv zero-pad
      for (int c = tid; c < 1280; c += 256) Urow[c] = (_Float16)0.f;
      continue;
    }
    float ty = (uy + 0.5f) * (1.0f / 32.0f) - 0.5f;
    float tx = (ux + 0.5f) * (1.0f / 32.0f) - 0.5f;
    int iy0 = (int)floorf(ty), ix0 = (int)floorf(tx);
    float fy = ty - (float)iy0, fx = tx - (float)ix0;
    int iy0c = min(max(iy0, 0), 7), iy1c = min(max(iy0 + 1, 0), 7);
    int ix0c = min(max(ix0, 0), 7), ix1c = min(max(ix0 + 1, 0), 7);
    const _Float16* p00 = featsH + (iy0c * 8 + ix0c) * 1280;
    const _Float16* p01 = featsH + (iy0c * 8 + ix1c) * 1280;
    const _Float16* p10 = featsH + (iy1c * 8 + ix0c) * 1280;
    const _Float16* p11 = featsH + (iy1c * 8 + ix1c) * 1280;
    float w00 = (1.f - fy) * (1.f - fx), w01 = (1.f - fy) * fx;
    float w10 = fy * (1.f - fx),        w11 = fy * fx;
    for (int c = tid; c < 1280; c += 256) {
      float v = w00 * (float)p00[c] + w01 * (float)p01[c] +
                w10 * (float)p10[c] + w11 * (float)p11[c];
      Urow[c] = (_Float16)v;
    }
  }
  __syncthreads();

  const int lane  = tid & 31;
  const int wave  = tid >> 5;
  const int msub  = wave & 3;              // 16-pixel group
  const int npair = wave >> 2;             // 64-channel half of the 128 tile
  const int l16   = lane & 15;
  const int hi    = lane >> 4;             // K-half select per WMMA layout
  const int p_local = msub * 16 + l16;
  const int py = p_local >> 3, px = p_local & 7;
  const int posbase = py * 10 + px;
  const int rot = (l16 >> 2) & 3;          // B row swizzle phase

  // per-thread staging addressing: 2 x b128 groups (128 rows x 32 k = 8KB)
  const int g0 = tid, g1 = tid + 256;
  const int sn0 = g0 >> 2, sc0 = g0 & 3;
  const int sn1 = g1 >> 2, sc1 = g1 & 3;
  const int d0 = sn0 * 32 + (((sc0 + ((sn0 >> 2) & 3)) & 3) << 3);
  const int d1 = sn1 * 32 + (((sc1 + ((sn1 >> 2) & 3)) & 3) << 3);

  float partial[8];
#pragma unroll
  for (int i = 0; i < 8; ++i) partial[i] = 0.f;

  for (int nt = 0; nt < 5; ++nt) {
    const _Float16* wnt = wt + (size_t)nt * 128 * 32;    // + ch * 640 * 32

    v8f acc[4];
#pragma unroll
    for (int j = 0; j < 4; ++j) acc[j] = (v8f){0.f,0.f,0.f,0.f,0.f,0.f,0.f,0.f};

#if SEG_ASYNC_LDS
    // prologue: async-copy chunk 0 -> Bs0
    __builtin_amdgcn_global_load_async_to_lds_b128(
        (as1_b128)(wnt + (size_t)g0 * 8), (as3_b128)(Bs0 + d0), 0, 0);
    __builtin_amdgcn_global_load_async_to_lds_b128(
        (as1_b128)(wnt + (size_t)g1 * 8), (as3_b128)(Bs0 + d1), 0, 0);
#else
    v8h r0 = *(const v8h*)(wnt + (size_t)g0 * 8);
    v8h r1 = *(const v8h*)(wnt + (size_t)g1 * 8);
#endif

    for (int ch = 0; ch < 360; ++ch) {
      _Float16* Bcur = (ch & 1) ? Bs1 : Bs0;
      _Float16* Bnxt = (ch & 1) ? Bs0 : Bs1;
      const _Float16* wch = wnt + (size_t)(ch + 1) * 640 * 32;

#if SEG_ASYNC_LDS
      __builtin_amdgcn_s_wait_asynccnt(0);   // Bcur copy landed in LDS
      __syncthreads();                       // everyone done reading Bnxt
      if (ch + 1 < 360) {
        __builtin_amdgcn_global_load_async_to_lds_b128(
            (as1_b128)(wch + (size_t)g0 * 8), (as3_b128)(Bnxt + d0), 0, 0);
        __builtin_amdgcn_global_load_async_to_lds_b128(
            (as1_b128)(wch + (size_t)g1 * 8), (as3_b128)(Bnxt + d1), 0, 0);
      }
#else
      __syncthreads();                       // Bcur free to overwrite
      *(v8h*)(Bcur + d0) = r0;
      *(v8h*)(Bcur + d1) = r1;
      if (ch + 1 < 360) {
        __builtin_prefetch(wch + (size_t)640 * 32, 0, 0);
        r0 = *(const v8h*)(wch + (size_t)g0 * 8);
        r1 = *(const v8h*)(wch + (size_t)g1 * 8);
      }
      __syncthreads();                       // staged tile visible
#endif

      // uniform chunk decode
      int kk9 = ch / 40;
      int cbase = (ch - kk9 * 40) * 32;
      int q = kk9 / 3;
      int off9 = q * 10 + (kk9 - q * 3);     // (ky)*10 + kx

      // A fragment (16-bit A 16x32: lanes<16 K0-7/16-23, lanes>=16 K8-15/24-31)
      const _Float16* ap = U + (size_t)(posbase + off9) * USTRIDE + cbase + hi * 8;
      v8h alo = *(const v8h*)(ap);
      v8h ahi = *(const v8h*)(ap + 16);
      v16h a = __builtin_shufflevector(alo, ahi, 0, 1, 2, 3, 4, 5, 6, 7,
                                       8, 9, 10, 11, 12, 13, 14, 15);

      // B fragments from swizzled rows
      const _Float16* brow = Bcur + (npair * 64 + l16) * 32;
      const int lo_off = ((hi + rot) & 3) << 3;
      const int hi_off = ((hi + 2 + rot) & 3) << 3;
#pragma unroll
      for (int j = 0; j < 4; ++j) {
        const _Float16* bp = brow + j * 16 * 32;
        v8h blo = *(const v8h*)(bp + lo_off);
        v8h bhi = *(const v8h*)(bp + hi_off);
        v16h bfr = __builtin_shufflevector(blo, bhi, 0, 1, 2, 3, 4, 5, 6, 7,
                                           8, 9, 10, 11, 12, 13, 14, 15);
        acc[j] = __builtin_amdgcn_wmma_f32_16x16x32_f16(false, a, false, bfr,
                                                        (short)0, acc[j], false, false);
      }
    }

    // epilogue: BN + ReLU + fused dot with seg_w2 (reduce N in-register)
    {
      const int nb = nt * 128 + npair * 64 + l16;
      float s[4], h[4], qq[4];
#pragma unroll
      for (int j = 0; j < 4; ++j) {
        int n = nb + j * 16;
        s[j] = bn_scale[n]; h[j] = bn_shift[n]; qq[j] = w2[n];
      }
#pragma unroll
      for (int i = 0; i < 8; ++i) {
        float contrib = 0.f;
#pragma unroll
        for (int j = 0; j < 4; ++j)
          contrib += fmaxf(acc[j][i] * s[j] + h[j], 0.f) * qq[j];
        contrib += __shfl_xor(contrib, 1, 32);
        contrib += __shfl_xor(contrib, 2, 32);
        contrib += __shfl_xor(contrib, 4, 32);
        contrib += __shfl_xor(contrib, 8, 32);
        partial[i] += contrib;               // lane-uniform within 16-half
      }
    }
  }

  // combine the two N-pair waves per pixel (one ds_add_f32 each -> commutative)
  if (l16 == 0) {
#pragma unroll
    for (int i = 0; i < 8; ++i) {
      int p = msub * 16 + hi * 8 + i;        // C layout: VGPR i -> M=i / i+8
      atomicAdd(&seg_acc[p], partial[i]);
    }
  }
  __syncthreads();
  if (tid < 64) {
    int pyo = tid >> 3, pxo = tid & 7;
    seg_out[(Y0 + pyo) * 256 + (X0 + pxo)] = seg_acc[tid] + b2[0];
  }
}

// --------------------------------- host ------------------------------------

extern "C" void kernel_launch(void* const* d_in, const int* in_sizes, int n_in,
                              void* d_out, int out_size, void* d_ws, size_t ws_size,
                              hipStream_t stream) {
  (void)in_sizes; (void)out_size; (void)ws_size;
  if (n_in < 166) return;

  const float* x = (const float*)d_in[0];

  // --- decode params: pytree flatten, sorted dict keys, bn = [scale, shift]
  struct BP {
    const float *dbn0, *dbn1, *dw, *ebn0, *ebn1, *ew, *pbn0, *pbn1, *pw;
    int cin, hid, cout, stride, expand;
  };
  BP bp[17];
  int pi = 1;
  {
    const int CFGt[7] = {1, 6, 6, 6, 6, 6, 6};
    const int CFGc[7] = {16, 24, 32, 64, 96, 160, 320};
    const int CFGn[7] = {1, 2, 3, 4, 3, 3, 1};
    const int CFGs[7] = {1, 2, 2, 2, 1, 2, 1};
    int cin = 32, bi = 0;
    for (int g = 0; g < 7; ++g)
      for (int i = 0; i < CFGn[g]; ++i, ++bi) {
        BP& B = bp[bi];
        B.cin = cin; B.hid = cin * CFGt[g]; B.cout = CFGc[g];
        B.stride = (i == 0) ? CFGs[g] : 1; B.expand = (CFGt[g] != 1);
        B.dbn0 = (const float*)d_in[pi + 0];
        B.dbn1 = (const float*)d_in[pi + 1];
        B.dw   = (const float*)d_in[pi + 2];
        if (B.expand) {
          B.ebn0 = (const float*)d_in[pi + 3];
          B.ebn1 = (const float*)d_in[pi + 4];
          B.ew   = (const float*)d_in[pi + 5];
          B.pbn0 = (const float*)d_in[pi + 6];
          B.pbn1 = (const float*)d_in[pi + 7];
          B.pw   = (const float*)d_in[pi + 8];
          pi += 9;
        } else {
          B.ebn0 = B.ebn1 = B.ew = nullptr;
          B.pbn0 = (const float*)d_in[pi + 3];
          B.pbn1 = (const float*)d_in[pi + 4];
          B.pw   = (const float*)d_in[pi + 5];
          pi += 6;
        }
        cin = CFGc[g];
      }
  }
  const float* head_bn0 = (const float*)d_in[pi++];
  const float* head_bn1 = (const float*)d_in[pi++];
  const float* head_w   = (const float*)d_in[pi++];
  const float* hm_b     = (const float*)d_in[pi++];
  const float* hm_w     = (const float*)d_in[pi++];
  const float* paf_b    = (const float*)d_in[pi++];
  const float* paf_w    = (const float*)d_in[pi++];
  const float* seg_b2   = (const float*)d_in[pi++];
  const float* seg_bn0  = (const float*)d_in[pi++];
  const float* seg_bn1  = (const float*)d_in[pi++];
  const float* seg_w1   = (const float*)d_in[pi++];
  const float* seg_w2   = (const float*)d_in[pi++];
  const float* stem_bn0 = (const float*)d_in[pi++];
  const float* stem_bn1 = (const float*)d_in[pi++];
  const float* stem_w   = (const float*)d_in[pi++];

  // --- workspace carve
  char* wsb = (char*)d_ws;
  size_t off = 0;
  auto carve = [&](size_t bytes) -> void* {
    void* p = wsb + off;
    off += bytes; off = (off + 255) & ~(size_t)255;
    return p;
  };
  const size_t ACT = (size_t)2 * 1024 * 1024 * sizeof(float);
  float* buf[4];
  for (int i = 0; i < 4; ++i) buf[i] = (float*)carve(ACT);
  float*    feats  = (float*)carve((size_t)1280 * 64 * sizeof(float));
  _Float16* featsH = (_Float16*)carve((size_t)64 * 1280 * sizeof(_Float16));
  _Float16* wt     = (_Float16*)carve((size_t)9 * 40 * 640 * 32 * sizeof(_Float16));

  float* d_seg   = (float*)d_out;                 // 65536
  float* d_hm    = d_seg + 65536;                 // 21*64
  float* d_paf   = d_hm + 21 * 64;                // 38*64
  float* d_peaks = d_paf + 38 * 64;               // 21*64

  // --- independent weight transform for the WMMA kernel
  {
    size_t n = (size_t)9 * 40 * 640 * 32;
    prep_wt_kernel<<<(unsigned)((n + 255) / 256), 256, 0, stream>>>(seg_w1, wt);
  }

  // --- stem
  stem_kernel<<<(32 * 128 * 128 + 255) / 256, 256, 0, stream>>>(x, stem_w, stem_bn0,
                                                                stem_bn1, buf[0]);
  // --- inverted-residual blocks
  float* y = buf[0];
  float* ynext = buf[1];
  int H = 128;
  for (int b = 0; b < 17; ++b) {
    const BP& B = bp[b];
    int Hout = (B.stride == 2) ? H / 2 : H;
    const float* z = y;
    int Cz = B.cin;
    if (B.expand) {
      int n = B.hid * H * H;
      conv1x1_kernel<<<(n + 255) / 256, 256, 0, stream>>>(y, B.ew, B.ebn0, B.ebn1,
                                                          nullptr, buf[2], B.cin,
                                                          B.hid, H * H, 1);
      z = buf[2]; Cz = B.hid;
    }
    {
      int n = Cz * Hout * Hout;
      dwconv_kernel<<<(n + 255) / 256, 256, 0, stream>>>(z, B.dw, B.dbn0, B.dbn1,
                                                         buf[3], Cz, H, Hout, B.stride);
    }
    {
      int n = B.cout * Hout * Hout;
      const float* res = (B.stride == 1 && B.cin == B.cout) ? y : nullptr;
      conv1x1_kernel<<<(n + 255) / 256, 256, 0, stream>>>(buf[3], B.pw, B.pbn0, B.pbn1,
                                                          res, ynext, Cz, B.cout,
                                                          Hout * Hout, 0);
    }
    float* t = y; y = ynext; ynext = t;
    H = Hout;
  }

  // --- head 1x1 -> feats fp32 (CHW), then fp16 NHWC mirror
  conv1x1_kernel<<<(1280 * 64 + 255) / 256, 256, 0, stream>>>(y, head_w, head_bn0,
                                                              head_bn1, nullptr, feats,
                                                              320, 1280, 64, 1);
  feats_half_kernel<<<(64 * 1280 + 255) / 256, 256, 0, stream>>>(feats, featsH);

  // --- hm / paf / peaks
  proj_kernel<<<(21 * 64 + 255) / 256, 256, 0, stream>>>(feats, hm_w, hm_b, d_hm, 21);
  proj_kernel<<<(38 * 64 + 255) / 256, 256, 0, stream>>>(feats, paf_w, paf_b, d_paf, 38);
  peaks_kernel<<<(21 * 64 + 255) / 256, 256, 0, stream>>>(d_hm, d_peaks);

  // --- seg branch: WMMA implicit GEMM, 1024 pixel tiles, ~274KB dynamic LDS
  seg_wmma_kernel<<<1024, 256, SEG_SMEM_BYTES, stream>>>(featsH, wt, seg_bn0, seg_bn1,
                                                         seg_w2, seg_b2, d_seg);
}
